// MaskedLineParamSSL_9577777070279
// MI455X (gfx1250) — matrix operand order repaired and checked by reading, not compiled, not run on hardware
//
#include <hip/hip_runtime.h>

#define NN 50000
#define EE 800000
#define HH 256
#define LL 4
#define NMASK 120000

typedef __attribute__((ext_vector_type(16))) __bf16 v16bf;
typedef __attribute__((ext_vector_type(8)))  float  v8f;

union FragBF { v16bf v; unsigned int u[8]; };

__device__ __forceinline__ unsigned short f32_to_bf16(float f) {
  unsigned int u = __float_as_uint(f);
  u += 0x7FFFu + ((u >> 16) & 1u);           // round-to-nearest-even
  return (unsigned short)(u >> 16);
}
__device__ __forceinline__ float bf16_to_f32(unsigned short s) {
  return __uint_as_float(((unsigned int)s) << 16);
}
__device__ __forceinline__ v8f wmma_bf16(const FragBF& a, const FragBF& b, v8f c) {
  return __builtin_amdgcn_wmma_f32_16x16x32_bf16(false, a.v, false, b.v,
                                                 (short)0, c, false, false);
}

// ---------------- prep kernels ----------------

__global__ __launch_bounds__(256) void pack_weights_k(
    const float* __restrict__ Wm, const float* __restrict__ Wu,
    const float* __restrict__ W1, unsigned short* __restrict__ wm,
    unsigned short* __restrict__ wu, unsigned short* __restrict__ w1) {
  int i = blockIdx.x * 256 + threadIdx.x;
  const int nWm = LL * 512 * HH;           // 524288
  const int nWu = LL * HH * HH;            // 262144
  const int nW1 = 512 * HH;                // 131072
  if (i < nWm) wm[i] = f32_to_bf16(Wm[i]);
  else if (i < nWm + nWu) wu[i - nWm] = f32_to_bf16(Wu[i - nWm]);
  else if (i < nWm + nWu + nW1) w1[i - nWm - nWu] = f32_to_bf16(W1[i - nWm - nWu]);
}

__global__ __launch_bounds__(256) void node_embed_k(
    const float* __restrict__ x, const float* __restrict__ Wn,
    const float* __restrict__ bn, float* __restrict__ h32,
    unsigned short* __restrict__ h16) {
  int t = blockIdx.x * 256 + threadIdx.x;
  if (t >= NN * HH) return;
  int i = t >> 8, j = t & 255;
  float v = bn[j] + x[i * 3 + 0] * Wn[0 * HH + j]
                  + x[i * 3 + 1] * Wn[1 * HH + j]
                  + x[i * 3 + 2] * Wn[2 * HH + j];
  v = v > 0.f ? v : 0.f;
  h32[t] = v;
  h16[t] = f32_to_bf16(v);
}

__global__ __launch_bounds__(256) void mask_attr_k(
    const float* __restrict__ edge_attr, const unsigned char* __restrict__ use_token,
    const unsigned char* __restrict__ use_rand, const float* __restrict__ rand_edge,
    const float* __restrict__ token, float* __restrict__ masked) {
  int e = blockIdx.x * 256 + threadIdx.x;
  if (e >= EE) return;
  bool ut = use_token[e] != 0;
  bool ur = use_rand[e] != 0;
#pragma unroll
  for (int c = 0; c < 2; ++c) {
    float v = ut ? token[c] : (ur ? rand_edge[e * 2 + c] : edge_attr[e * 2 + c]);
    masked[e * 2 + c] = v;
  }
}

// ---------------- message GEMM + fused segment-sum scatter ----------------
// Block: 256 threads (8 waves), 64 edges, full 256 output cols.
// A = [h[src] | e_recomputed]  (64x512 bf16, staged 64x32 per K-chunk)
// B = Wm[l]                    (512x256 bf16, staged transposed 32x256)

__global__ __launch_bounds__(256) void msg_scatter_k(
    const unsigned short* __restrict__ h16, const float* __restrict__ masked,
    const float* __restrict__ We, const float* __restrict__ be,
    const unsigned short* __restrict__ Wl, const float* __restrict__ bl,
    const int* __restrict__ ei, float* __restrict__ agg) {
  __shared__ unsigned int ldsA[64 * 16];    // 64 rows x 32 bf16
  __shared__ unsigned int ldsB[256 * 16];   // 256 cols x 32 bf16 (K-transposed)
  __shared__ int ldsSrc[64];
  __shared__ int ldsDst[64];
  const int tid = threadIdx.x;
  const int ebase = blockIdx.x * 64;
  if (tid < 64) {
    ldsSrc[tid] = ei[ebase + tid];
    ldsDst[tid] = ei[EE + ebase + tid];
  }
  __syncthreads();
  const int lane = tid & 31, wv = tid >> 5, hf = lane >> 4, nl = lane & 15;
  unsigned short* sA = (unsigned short*)ldsA;
  unsigned short* sB = (unsigned short*)ldsB;

  v8f acc[4][2];
#pragma unroll
  for (int mt = 0; mt < 4; ++mt)
#pragma unroll
    for (int j = 0; j < 2; ++j)
#pragma unroll
      for (int r = 0; r < 8; ++r) acc[mt][j][r] = 0.f;

  for (int kc = 0; kc < 16; ++kc) {
    __syncthreads();
    // stage A: 64x32 bf16
#pragma unroll
    for (int it = 0; it < 8; ++it) {
      int idx = tid + it * 256;
      int row = idx >> 5, k = idx & 31;
      int gk = kc * 32 + k;
      unsigned short val;
      if (gk < HH) {
        val = h16[ldsSrc[row] * HH + gk];
      } else {
        int j = gk - HH;
        int e = ebase + row;
        float f = masked[e * 2 + 0] * We[j] + masked[e * 2 + 1] * We[HH + j] + be[j];
        f = f > 0.f ? f : 0.f;
        val = f32_to_bf16(f);
      }
      sA[idx] = val;
    }
    // stage B transposed: col n = tid
    {
      const unsigned short* srcB = Wl + (kc * 32) * HH + tid;
#pragma unroll
      for (int kk = 0; kk < 32; ++kk) sB[tid * 32 + kk] = srcB[kk * HH];
    }
    __syncthreads();

    FragBF a[4], b[2];
#pragma unroll
    for (int j = 0; j < 2; ++j) {
      int n = (2 * wv + j) * 16 + nl;
#pragma unroll
      for (int vv = 0; vv < 8; ++vv) b[j].u[vv] = ldsB[n * 16 + hf * 8 + vv];
    }
#pragma unroll
    for (int mt = 0; mt < 4; ++mt) {
      int m = mt * 16 + nl;
#pragma unroll
      for (int vv = 0; vv < 8; ++vv)
        a[mt].u[vv] = ldsA[m * 16 + hf * 8 + vv];
    }
#pragma unroll
    for (int mt = 0; mt < 4; ++mt)
#pragma unroll
      for (int j = 0; j < 2; ++j) acc[mt][j] = wmma_bf16(a[mt], b[j], acc[mt][j]);
  }

  // epilogue: bias + relu + atomic scatter-add (segment_sum) into L2-resident agg
#pragma unroll
  for (int j = 0; j < 2; ++j) {
    int n = (2 * wv + j) * 16 + nl;
    float bias = bl[n];
#pragma unroll
    for (int mt = 0; mt < 4; ++mt) {
#pragma unroll
      for (int r = 0; r < 8; ++r) {
        int mloc = mt * 16 + r + hf * 8;
        float v = acc[mt][j][r] + bias;
        v = v > 0.f ? v : 0.f;
        atomicAdd(&agg[ldsDst[mloc] * HH + n], v);
      }
    }
  }
}

// ---------------- node update GEMM: h = relu(h + agg @ Wu[l] + bu[l]) ----------------

__global__ __launch_bounds__(256) void update_h_k(
    const float* __restrict__ agg, const unsigned short* __restrict__ Wl,
    const float* __restrict__ bl, float* __restrict__ h32,
    unsigned short* __restrict__ h16) {
  __shared__ unsigned int ldsA[64 * 16];
  __shared__ unsigned int ldsB[256 * 16];
  const int tid = threadIdx.x;
  const int rbase = blockIdx.x * 64;
  const int lane = tid & 31, wv = tid >> 5, hf = lane >> 4, nl = lane & 15;
  unsigned short* sA = (unsigned short*)ldsA;
  unsigned short* sB = (unsigned short*)ldsB;

  v8f acc[4][2];
#pragma unroll
  for (int mt = 0; mt < 4; ++mt)
#pragma unroll
    for (int j = 0; j < 2; ++j)
#pragma unroll
      for (int r = 0; r < 8; ++r) acc[mt][j][r] = 0.f;

  for (int kc = 0; kc < 8; ++kc) {
    __syncthreads();
#pragma unroll
    for (int it = 0; it < 8; ++it) {
      int idx = tid + it * 256;
      int row = idx >> 5, k = idx & 31;
      int grow = rbase + row, gk = kc * 32 + k;
      float f = (grow < NN) ? agg[grow * HH + gk] : 0.f;
      sA[idx] = f32_to_bf16(f);
    }
    const unsigned short* srcB = Wl + (kc * 32) * HH + tid;
#pragma unroll
    for (int kk = 0; kk < 32; ++kk) sB[tid * 32 + kk] = srcB[kk * HH];
    __syncthreads();

    FragBF a[4], b[2];
#pragma unroll
    for (int j = 0; j < 2; ++j) {
      int n = (2 * wv + j) * 16 + nl;
#pragma unroll
      for (int vv = 0; vv < 8; ++vv) b[j].u[vv] = ldsB[n * 16 + hf * 8 + vv];
    }
#pragma unroll
    for (int mt = 0; mt < 4; ++mt) {
      int m = mt * 16 + nl;
#pragma unroll
      for (int vv = 0; vv < 8; ++vv)
        a[mt].u[vv] = ldsA[m * 16 + hf * 8 + vv];
    }
#pragma unroll
    for (int mt = 0; mt < 4; ++mt)
#pragma unroll
      for (int j = 0; j < 2; ++j) acc[mt][j] = wmma_bf16(a[mt], b[j], acc[mt][j]);
  }

#pragma unroll
  for (int j = 0; j < 2; ++j) {
    int n = (2 * wv + j) * 16 + nl;
    float bias = bl[n];
#pragma unroll
    for (int mt = 0; mt < 4; ++mt) {
#pragma unroll
      for (int r = 0; r < 8; ++r) {
        int grow = rbase + mt * 16 + r + hf * 8;
        if (grow < NN) {
          int o = grow * HH + n;
          float v = h32[o] + acc[mt][j][r] + bias;
          v = v > 0.f ? v : 0.f;
          h32[o] = v;
          h16[o] = f32_to_bf16(v);
        }
      }
    }
  }
}

// ---------------- reconstruction: relu([h[src]|h[dst]] @ W1 + b1) @ W2 + b2 ----------------

__global__ __launch_bounds__(256) void recon_k(
    const unsigned short* __restrict__ h16, const unsigned short* __restrict__ W1b,
    const float* __restrict__ b1, const float* __restrict__ W2,
    const float* __restrict__ b2, const int* __restrict__ ei,
    float* __restrict__ out) {
  __shared__ unsigned int ldsA[64 * 16];
  __shared__ unsigned int ldsB[256 * 16];
  __shared__ unsigned short ldsT[64 * HH];   // 64x256 bf16 intermediate
  __shared__ int ldsSrc[64];
  __shared__ int ldsDst[64];
  const int tid = threadIdx.x;
  const int ebase = blockIdx.x * 64;
  if (tid < 64) {
    ldsSrc[tid] = ei[ebase + tid];
    ldsDst[tid] = ei[EE + ebase + tid];
  }
  __syncthreads();
  const int lane = tid & 31, wv = tid >> 5, hf = lane >> 4, nl = lane & 15;
  unsigned short* sA = (unsigned short*)ldsA;
  unsigned short* sB = (unsigned short*)ldsB;

  v8f acc[4][2];
#pragma unroll
  for (int mt = 0; mt < 4; ++mt)
#pragma unroll
    for (int j = 0; j < 2; ++j)
#pragma unroll
      for (int r = 0; r < 8; ++r) acc[mt][j][r] = 0.f;

  for (int kc = 0; kc < 16; ++kc) {
    __syncthreads();
#pragma unroll
    for (int it = 0; it < 8; ++it) {
      int idx = tid + it * 256;
      int row = idx >> 5, k = idx & 31;
      int gk = kc * 32 + k;
      unsigned short val = (gk < HH) ? h16[ldsSrc[row] * HH + gk]
                                     : h16[ldsDst[row] * HH + (gk - HH)];
      sA[idx] = val;
    }
    const unsigned short* srcB = W1b + (kc * 32) * HH + tid;
#pragma unroll
    for (int kk = 0; kk < 32; ++kk) sB[tid * 32 + kk] = srcB[kk * HH];
    __syncthreads();

    FragBF a[4], b[2];
#pragma unroll
    for (int j = 0; j < 2; ++j) {
      int n = (2 * wv + j) * 16 + nl;
#pragma unroll
      for (int vv = 0; vv < 8; ++vv) b[j].u[vv] = ldsB[n * 16 + hf * 8 + vv];
    }
#pragma unroll
    for (int mt = 0; mt < 4; ++mt) {
      int m = mt * 16 + nl;
#pragma unroll
      for (int vv = 0; vv < 8; ++vv)
        a[mt].u[vv] = ldsA[m * 16 + hf * 8 + vv];
    }
#pragma unroll
    for (int mt = 0; mt < 4; ++mt)
#pragma unroll
      for (int j = 0; j < 2; ++j) acc[mt][j] = wmma_bf16(a[mt], b[j], acc[mt][j]);
  }

  // stash relu(GEMM1 + b1) tile in LDS
#pragma unroll
  for (int j = 0; j < 2; ++j) {
    int n = (2 * wv + j) * 16 + nl;
    float bias = b1[n];
#pragma unroll
    for (int mt = 0; mt < 4; ++mt) {
#pragma unroll
      for (int r = 0; r < 8; ++r) {
        int mloc = mt * 16 + r + hf * 8;
        float v = acc[mt][j][r] + bias;
        v = v > 0.f ? v : 0.f;
        ldsT[mloc * HH + n] = f32_to_bf16(v);
      }
    }
  }
  __syncthreads();

  // tiny head: (64x256) @ (256x2) + b2
  if (tid < 128) {
    int el = tid >> 1, c = tid & 1;
    float s = b2[c];
#pragma unroll 4
    for (int j = 0; j < HH; ++j) s += bf16_to_f32(ldsT[el * HH + j]) * W2[j * 2 + c];
    out[1 + (ebase + el) * 2 + c] = s;
  }
}

// ---------------- masked MSE loss ----------------

__global__ __launch_bounds__(256) void loss_k(
    const float* __restrict__ outbuf, const float* __restrict__ edge_attr,
    const int* __restrict__ mask_idx, float* __restrict__ loss) {
  __shared__ float red[256];
  int tid = threadIdx.x;
  int i = blockIdx.x * 256 + tid;
  float s = 0.f;
  if (i < NMASK) {
    int e = mask_idx[i];
    float d0 = outbuf[1 + e * 2 + 0] - edge_attr[e * 2 + 0];
    float d1 = outbuf[1 + e * 2 + 1] - edge_attr[e * 2 + 1];
    s = d0 * d0 + d1 * d1;
  }
  red[tid] = s;
  __syncthreads();
  for (int st = 128; st > 0; st >>= 1) {
    if (tid < st) red[tid] += red[tid + st];
    __syncthreads();
  }
  if (tid == 0) atomicAdd(loss, red[0] * (1.f / (NMASK * 2.f)));
}

// ---------------- host launch ----------------

extern "C" void kernel_launch(void* const* d_in, const int* in_sizes, int n_in,
                              void* d_out, int out_size, void* d_ws, size_t ws_size,
                              hipStream_t stream) {
  (void)in_sizes; (void)n_in; (void)out_size; (void)ws_size;
  const float* x          = (const float*)d_in[0];
  const int*   ei         = (const int*)d_in[1];
  const float* edge_attr  = (const float*)d_in[2];
  const int*   mask_idx   = (const int*)d_in[3];
  const unsigned char* use_token = (const unsigned char*)d_in[4];
  const unsigned char* use_rand  = (const unsigned char*)d_in[5];
  const float* rand_edge  = (const float*)d_in[6];
  const float* token      = (const float*)d_in[7];
  const float* Wn = (const float*)d_in[8];
  const float* bn = (const float*)d_in[9];
  const float* We = (const float*)d_in[10];
  const float* be = (const float*)d_in[11];
  const float* Wm = (const float*)d_in[12];
  const float* bm = (const float*)d_in[13];
  const float* Wu = (const float*)d_in[14];
  const float* bu = (const float*)d_in[15];
  const float* W1 = (const float*)d_in[16];
  const float* b1 = (const float*)d_in[17];
  const float* W2 = (const float*)d_in[18];
  const float* b2 = (const float*)d_in[19];
  float* out = (float*)d_out;

  char* ws = (char*)d_ws;
  float*          h32    = (float*)(ws + 0);                  // 51,200,000 B
  unsigned short* h16    = (unsigned short*)(ws + 51200000);  // 25,600,000 B
  float*          agg    = (float*)(ws + 76800000);           // 51,200,000 B
  float*          masked = (float*)(ws + 128000000);          //  6,400,000 B
  unsigned short* wm_b   = (unsigned short*)(ws + 134400000); //  1,048,576 B
  unsigned short* wu_b   = (unsigned short*)(ws + 135448576); //    524,288 B
  unsigned short* w1_b   = (unsigned short*)(ws + 135972864); //    262,144 B

  pack_weights_k<<<(917504 + 255) / 256, 256, 0, stream>>>(Wm, Wu, W1, wm_b, wu_b, w1_b);
  node_embed_k<<<(NN * HH + 255) / 256, 256, 0, stream>>>(x, Wn, bn, h32, h16);
  mask_attr_k<<<(EE + 255) / 256, 256, 0, stream>>>(edge_attr, use_token, use_rand,
                                                    rand_edge, token, masked);
  for (int l = 0; l < LL; ++l) {
    hipMemsetAsync(agg, 0, (size_t)NN * HH * sizeof(float), stream);
    msg_scatter_k<<<EE / 64, 256, 0, stream>>>(h16, masked, We, be,
                                               wm_b + l * 512 * HH, bm + l * HH, ei, agg);
    update_h_k<<<(NN + 63) / 64, 256, 0, stream>>>(agg, wu_b + l * HH * HH,
                                                   bu + l * HH, h32, h16);
  }
  recon_k<<<EE / 64, 256, 0, stream>>>(h16, w1_b, b1, W2, b2, ei, out);
  hipMemsetAsync(out, 0, sizeof(float), stream);
  loss_k<<<(NMASK + 255) / 256, 256, 0, stream>>>(out, edge_attr, mask_idx, out);
}
